// PredictionModel_41291815584476
// MI455X (gfx1250) — matrix operand order repaired and checked by reading, not compiled, not run on hardware
//
#include <hip/hip_runtime.h>
#include <hip/hip_bf16.h>

// Problem constants (from reference)
#define B_     128
#define T_     256
#define F_     128
#define NCAT_  32
#define CARD_  16
#define E_     8
#define H_     512
#define IN_    352          // F + NCAT*(E-1)
#define G4_    2048         // 4*H
#define NCONT_ 96
#define MT_    (B_ * T_)    // 32768 rows for the input projection

typedef __attribute__((ext_vector_type(16))) __bf16          v16bf;
typedef __attribute__((ext_vector_type(16))) unsigned short  v16u;
typedef __attribute__((ext_vector_type(8)))  unsigned short  v8u;
typedef __attribute__((ext_vector_type(8)))  float           v8f;

__device__ __forceinline__ unsigned short f2bf(float f) {
  unsigned int u = __builtin_bit_cast(unsigned int, f);
  u = (u + 0x7FFFu + ((u >> 16) & 1u)) >> 16;   // round-to-nearest-even
  return (unsigned short)u;
}
__device__ __forceinline__ float sigf(float x) { return 1.0f / (1.0f + __expf(-x)); }

// A-matrix tile load (16x32 bf16): lane<16 holds K {0..15}, lane>=16 holds K {16..31}
__device__ __forceinline__ v16u loadA(const unsigned short* p, int sel) {
  const unsigned short* q = p + sel * 16;
  v8u lo = *(const v8u*)q;
  v8u hi = *(const v8u*)(q + 8);
  v16u r;
#pragma unroll
  for (int i = 0; i < 8; ++i) { r[i] = lo[i]; r[i + 8] = hi[i]; }
  return r;
}

__device__ __forceinline__ v8f wmma_bf16(v16u a, v16u b, v8f c) {
  return __builtin_amdgcn_wmma_f32_16x16x32_bf16(
      false, __builtin_bit_cast(v16bf, a),
      false, __builtin_bit_cast(v16bf, b),
      (short)0, c, false, false);
}

// ---------------- setup kernels ----------------

// Build bf16 input matrix Xin[(t*B+b), 352] = concat(emb gather, cont) per group
__global__ void gather_kernel(const int* __restrict__ us, const float* __restrict__ ss,
                              const float* __restrict__ emb, unsigned short* __restrict__ XIN) {
  int idx = blockIdx.x * blockDim.x + threadIdx.x;
  if (idx >= MT_ * IN_) return;
  int r = idx / IN_;
  int k = idx - r * IN_;
  int t = r / B_;
  int b = r - t * B_;
  int g = k / 11;
  int j = k - g * 11;
  size_t base = ((size_t)b * T_ + t) * F_ + 4 * g;
  float val;
  if (j < 8) {
    int id = us[base];
    val = emb[(g * CARD_ + id) * E_ + j];
  } else {
    val = ss[base + (j - 7)];
  }
  XIN[idx] = f2bf(val);
}

// Winbf[n][k] = bf16(W_in[k][n])  (transpose to N-major rows of K)
__global__ void convert_win_kernel(const float* __restrict__ W_in, unsigned short* __restrict__ WB) {
  int idx = blockIdx.x * blockDim.x + threadIdx.x;
  if (idx >= H_ * IN_) return;
  int n = idx / IN_;
  int k = idx - n * IN_;
  WB[idx] = f2bf(W_in[k * H_ + n]);
}

// W0cat[n][k] = k<512 ? Wih0[n][k] : Whh0[n][k-512]   (2048 x 1024 bf16)
__global__ void convert_w0_kernel(const float* __restrict__ Wih0, const float* __restrict__ Whh0,
                                  unsigned short* __restrict__ W0) {
  int idx = blockIdx.x * blockDim.x + threadIdx.x;
  if (idx >= G4_ * 2 * H_) return;
  int n = idx / (2 * H_);
  int k = idx - n * (2 * H_);
  float v = (k < H_) ? Wih0[(size_t)n * H_ + k] : Whh0[(size_t)n * H_ + (k - H_)];
  W0[idx] = f2bf(v);
}

// W1sum = bf16(Wih1 + Whh1)   (2048 x 512 bf16)
__global__ void convert_w1_kernel(const float* __restrict__ Wih1, const float* __restrict__ Whh1,
                                  unsigned short* __restrict__ W1) {
  int idx = blockIdx.x * blockDim.x + threadIdx.x;
  if (idx >= G4_ * H_) return;
  W1[idx] = f2bf(Wih1[idx] + Whh1[idx]);
}

// b0 = bih0+bhh0, b1 = bih1+bhh1, zero h carry (bf16) and c carry (f32)
__global__ void setup_misc_kernel(const float* __restrict__ bih0, const float* __restrict__ bhh0,
                                  const float* __restrict__ bih1, const float* __restrict__ bhh1,
                                  float* __restrict__ B0, float* __restrict__ B1,
                                  unsigned short* __restrict__ HBF, float* __restrict__ CC) {
  int idx = blockIdx.x * blockDim.x + threadIdx.x;
  if (idx < G4_) { B0[idx] = bih0[idx] + bhh0[idx]; B1[idx] = bih1[idx] + bhh1[idx]; }
  if (idx < B_ * H_) { HBF[idx] = 0; CC[idx] = 0.0f; }
}

// ---------------- input projection GEMM (WMMA bf16) ----------------
// X[(t*B+b)][n] = bf16(relu( Xin row . Winbf row n + b_in[n] ))
__global__ __launch_bounds__(32) void gemm_in_kernel(const unsigned short* __restrict__ XIN,
                                                     const unsigned short* __restrict__ WB,
                                                     const float* __restrict__ b_in,
                                                     unsigned short* __restrict__ X) {
  const int lane = threadIdx.x & 31;
  const int sel = lane >> 4;
  const int l15 = lane & 15;
  const int n0 = blockIdx.x * 16;
  const size_t m0 = (size_t)blockIdx.y * 16;

  v8f acc = {};
  const unsigned short* arow = XIN + (m0 + l15) * IN_;
  const unsigned short* brow = WB + (size_t)(n0 + l15) * IN_;
#pragma unroll 11
  for (int kk = 0; kk < IN_; kk += 32) {
    v16u a = loadA(arow + kk, sel);
    v16u b = *(const v16u*)(brow + kk + sel * 16);
    acc = wmma_bf16(a, b, acc);
  }
  const float bn = b_in[n0 + l15];
#pragma unroll
  for (int v = 0; v < 8; ++v) {
    int m = v + sel * 8;
    float val = acc[v] + bn;
    val = fmaxf(val, 0.0f);
    X[(m0 + m) * H_ + (n0 + l15)] = f2bf(val);
  }
}

// ---------------- recurrent cells (fused WMMA GEMM + LSTM epilogue) ----------------
// Cell 0: gates = [x_t | h_prev] @ W0cat^T + b0 ; c0 = sig(f)*c + sig(i)*tanh(g); h0 = sig(o)*tanh(c0)
__global__ __launch_bounds__(32) void cell0_kernel(const unsigned short* __restrict__ Xt,
                                                   const unsigned short* __restrict__ Hbf,
                                                   const unsigned short* __restrict__ W0,
                                                   const float* __restrict__ b0,
                                                   const float* __restrict__ Cin,
                                                   float* __restrict__ Cout,
                                                   unsigned short* __restrict__ H0bf) {
  const int lane = threadIdx.x & 31;
  const int sel = lane >> 4;
  const int l15 = lane & 15;
  const int n0 = blockIdx.x * 16;   // hidden-column tile (0..511)
  const int m0 = blockIdx.y * 16;   // batch tile (0..127)

  v8f acc[4];
#pragma unroll
  for (int g = 0; g < 4; ++g) acc[g] = (v8f){};

  const int rowA = m0 + l15;
  const unsigned short* xrow = Xt + (size_t)rowA * H_;
  const unsigned short* hrow = Hbf + (size_t)rowA * H_;

  for (int kk = 0; kk < 2 * H_; kk += 32) {
    const unsigned short* asrc = (kk < H_) ? (xrow + kk) : (hrow + (kk - H_));
    v16u a = loadA(asrc, sel);
#pragma unroll
    for (int g = 0; g < 4; ++g) {
      const unsigned short* bp = W0 + (size_t)(n0 + g * H_ + l15) * (2 * H_) + kk + sel * 16;
      v16u b = *(const v16u*)bp;
      acc[g] = wmma_bf16(a, b, acc[g]);
    }
  }

  const int hidx = n0 + l15;
  const float bi = b0[hidx];
  const float bf_ = b0[hidx + H_];
  const float bg = b0[hidx + 2 * H_];
  const float bo = b0[hidx + 3 * H_];
#pragma unroll
  for (int v = 0; v < 8; ++v) {
    int row = m0 + v + sel * 8;
    float gi = acc[0][v] + bi;
    float gf = acc[1][v] + bf_;
    float gg = acc[2][v] + bg;
    float go = acc[3][v] + bo;
    float cold = Cin[row * H_ + hidx];
    float cn = sigf(gf) * cold + sigf(gi) * tanhf(gg);
    float h0 = sigf(go) * tanhf(cn);
    Cout[row * H_ + hidx] = cn;
    H0bf[row * H_ + hidx] = f2bf(h0);
  }
}

// Cell 1: x == h == h0, so gates = h0 @ (Wih1+Whh1)^T + (bih1+bhh1)
__global__ __launch_bounds__(32) void cell1_kernel(const unsigned short* __restrict__ H0bf,
                                                   const unsigned short* __restrict__ W1,
                                                   const float* __restrict__ b1,
                                                   const float* __restrict__ Cin,
                                                   float* __restrict__ Cout,
                                                   unsigned short* __restrict__ Hbf,
                                                   float* __restrict__ HF) {
  const int lane = threadIdx.x & 31;
  const int sel = lane >> 4;
  const int l15 = lane & 15;
  const int n0 = blockIdx.x * 16;
  const int m0 = blockIdx.y * 16;

  v8f acc[4];
#pragma unroll
  for (int g = 0; g < 4; ++g) acc[g] = (v8f){};

  const unsigned short* arow = H0bf + (size_t)(m0 + l15) * H_;

  for (int kk = 0; kk < H_; kk += 32) {
    v16u a = loadA(arow + kk, sel);
#pragma unroll
    for (int g = 0; g < 4; ++g) {
      const unsigned short* bp = W1 + (size_t)(n0 + g * H_ + l15) * H_ + kk + sel * 16;
      v16u b = *(const v16u*)bp;
      acc[g] = wmma_bf16(a, b, acc[g]);
    }
  }

  const int hidx = n0 + l15;
  const float bi = b1[hidx];
  const float bf_ = b1[hidx + H_];
  const float bg = b1[hidx + 2 * H_];
  const float bo = b1[hidx + 3 * H_];
#pragma unroll
  for (int v = 0; v < 8; ++v) {
    int row = m0 + v + sel * 8;
    float gi = acc[0][v] + bi;
    float gf = acc[1][v] + bf_;
    float gg = acc[2][v] + bg;
    float go = acc[3][v] + bo;
    float cold = Cin[row * H_ + hidx];
    float cn = sigf(gf) * cold + sigf(gi) * tanhf(gg);
    float hn = sigf(go) * tanhf(cn);
    Cout[row * H_ + hidx] = cn;
    Hbf[row * H_ + hidx] = f2bf(hn);
    HF[row * H_ + hidx] = hn;
  }
}

// ---------------- output heads ----------------
__global__ void head_cont_kernel(const float* __restrict__ HF, const float* __restrict__ W_out,
                                 const float* __restrict__ b_out, float* __restrict__ out) {
  int idx = blockIdx.x * blockDim.x + threadIdx.x;
  if (idx >= B_ * NCONT_) return;
  int b = idx / NCONT_;
  int n = idx - b * NCONT_;
  float s = b_out[n];
  const float* h = HF + b * H_;
  for (int k = 0; k < H_; ++k) s += h[k] * W_out[k * NCONT_ + n];
  out[idx] = s;
}

__global__ void head_cls_kernel(const float* __restrict__ HF, const float* __restrict__ W_cls,
                                const float* __restrict__ b_cls, float* __restrict__ out) {
  int idx = blockIdx.x * blockDim.x + threadIdx.x;
  if (idx >= B_ * NCAT_ * CARD_) return;
  int b = idx / (NCAT_ * CARD_);
  int rem = idx - b * (NCAT_ * CARD_);
  int kc = rem / CARD_;
  int c = rem - kc * CARD_;
  float s = b_cls[kc * CARD_ + c];
  const float* h = HF + b * H_;
  const float* w = W_cls + ((size_t)kc * H_) * CARD_ + c;
  for (int k = 0; k < H_; ++k) s += h[k] * w[(size_t)k * CARD_];
  out[idx] = s;
}

__global__ void copy_hc_kernel(const float* __restrict__ HF, const float* __restrict__ CC,
                               float* __restrict__ out) {
  int idx = blockIdx.x * blockDim.x + threadIdx.x;
  if (idx >= B_ * H_) return;
  out[idx] = HF[idx];
  out[B_ * H_ + idx] = CC[idx];
}

// ---------------- host launcher ----------------
extern "C" void kernel_launch(void* const* d_in, const int* in_sizes, int n_in,
                              void* d_out, int out_size, void* d_ws, size_t ws_size,
                              hipStream_t stream) {
  (void)in_sizes; (void)n_in; (void)out_size; (void)ws_size;
  const int*   us    = (const int*)  d_in[0];
  const float* ss    = (const float*)d_in[1];
  const float* emb   = (const float*)d_in[2];
  const float* W_in  = (const float*)d_in[3];
  const float* b_in  = (const float*)d_in[4];
  const float* Wih0  = (const float*)d_in[5];
  const float* Whh0  = (const float*)d_in[6];
  const float* bih0  = (const float*)d_in[7];
  const float* bhh0  = (const float*)d_in[8];
  const float* Wih1  = (const float*)d_in[9];
  const float* Whh1  = (const float*)d_in[10];
  const float* bih1  = (const float*)d_in[11];
  const float* bhh1  = (const float*)d_in[12];
  const float* W_out = (const float*)d_in[13];
  const float* b_out = (const float*)d_in[14];
  const float* W_cls = (const float*)d_in[15];
  const float* b_cls = (const float*)d_in[16];
  float* out = (float*)d_out;

  char* w = (char*)d_ws;
  auto carve = [&](size_t bytes) -> char* {
    char* p = w;
    w += (bytes + 255) & ~(size_t)255;
    return p;
  };
  unsigned short* XIN  = (unsigned short*)carve((size_t)MT_ * IN_ * 2);
  unsigned short* X    = (unsigned short*)carve((size_t)MT_ * H_ * 2);
  unsigned short* WINB = (unsigned short*)carve((size_t)H_ * IN_ * 2);
  unsigned short* W0   = (unsigned short*)carve((size_t)G4_ * 2 * H_ * 2);
  unsigned short* W1   = (unsigned short*)carve((size_t)G4_ * H_ * 2);
  float*          B0   = (float*)carve(G4_ * 4);
  float*          B1   = (float*)carve(G4_ * 4);
  unsigned short* HBF  = (unsigned short*)carve((size_t)B_ * H_ * 2);
  unsigned short* H0B  = (unsigned short*)carve((size_t)B_ * H_ * 2);
  float*          HF   = (float*)carve((size_t)B_ * H_ * 4);
  float*          CC   = (float*)carve((size_t)B_ * H_ * 4);
  float*          CM   = (float*)carve((size_t)B_ * H_ * 4);

  gather_kernel<<<(MT_ * IN_ + 255) / 256, 256, 0, stream>>>(us, ss, emb, XIN);
  convert_win_kernel<<<(H_ * IN_ + 255) / 256, 256, 0, stream>>>(W_in, WINB);
  convert_w0_kernel<<<(G4_ * 2 * H_) / 256, 256, 0, stream>>>(Wih0, Whh0, W0);
  convert_w1_kernel<<<(G4_ * H_) / 256, 256, 0, stream>>>(Wih1, Whh1, W1);
  setup_misc_kernel<<<(B_ * H_) / 256, 256, 0, stream>>>(bih0, bhh0, bih1, bhh1, B0, B1, HBF, CC);

  // x -> relu(x @ W_in + b_in), stored bf16 as (T,B,H)
  gemm_in_kernel<<<dim3(H_ / 16, MT_ / 16), 32, 0, stream>>>(XIN, WINB, b_in, X);

  // Sequential recurrence: 2 fused WMMA+activation kernels per step
  for (int t = 0; t < T_; ++t) {
    cell0_kernel<<<dim3(H_ / 16, B_ / 16), 32, 0, stream>>>(
        X + (size_t)t * B_ * H_, HBF, W0, B0, CC, CM, H0B);
    cell1_kernel<<<dim3(H_ / 16, B_ / 16), 32, 0, stream>>>(
        H0B, W1, B1, CM, CC, HBF, HF);
  }

  head_cont_kernel<<<(B_ * NCONT_ + 255) / 256, 256, 0, stream>>>(HF, W_out, b_out, out);
  head_cls_kernel<<<(B_ * NCAT_ * CARD_) / 256, 256, 0, stream>>>(
      HF, W_cls, b_cls, out + B_ * NCONT_);
  copy_hc_kernel<<<(B_ * H_) / 256, 256, 0, stream>>>(
      HF, CC, out + B_ * NCONT_ + B_ * NCAT_ * CARD_);
}
